// CustomLinear4Bit_10900626998011
// MI455X (gfx1250) — compile-verified, matrix operand order checked by reading, not run on hardware
//
#include <hip/hip_runtime.h>
#include <hip/hip_bf16.h>

typedef __attribute__((ext_vector_type(16))) _Float16 v16h;
typedef __attribute__((ext_vector_type(8)))  _Float16 v8h;
typedef __attribute__((ext_vector_type(4)))  _Float16 v4h;
typedef __attribute__((ext_vector_type(8)))  float    v8f;
typedef __attribute__((ext_vector_type(4)))  float    v4f;
typedef __attribute__((ext_vector_type(4)))  int      v4i;

#define M_TOK   8192      // 4 * 2048
#define K_DIM   4096
#define N_DIM   16384
#define GS      16
#define KGROUPS (K_DIM / GS)   // 256 groups per output row

#define BM 128
#define BN 128
#define BK 32
#define NKSTEPS (K_DIM / BK)   // 128
#define LDA 40   // halves per LDS row (32 + 8 pad): 80 B rows, 16B-aligned chunks
#define LDB 40
#define ATILE (BM * LDA)
#define BTILE (BN * LDB)

union F16x16 { v16h v; v8h h[2]; };
union WBuf   { v8h v[2]; _Float16 e[16]; };

__global__ __launch_bounds__(256)
void w4a16_wmma_gemm(const float* __restrict__ X,
                     const int*   __restrict__ QW,
                     const float* __restrict__ SC,
                     const float* __restrict__ BIAS,
                     float*       __restrict__ OUT) {
    // double-buffered staging tiles
    __shared__ _Float16 As[2 * ATILE];
    __shared__ _Float16 Bs[2 * BTILE];

    const int tid  = threadIdx.x;
    const int lane = tid & 31;
    const int wave = tid >> 5;
    const int hh   = lane >> 4;   // K-chunk select within fragment
    const int lr   = lane & 15;   // row/col within fragment

    const int m0 = blockIdx.y * BM;
    const int n0 = blockIdx.x * BN;

    // 8 waves: 2 along M x 4 along N; each wave = 64x32 output = 4x2 fragments
    const int wm = (wave & 1) * 64;
    const int wn = (wave >> 1) * 32;

    v8f acc[4][2];
#pragma unroll
    for (int mi = 0; mi < 4; ++mi)
#pragma unroll
        for (int ni = 0; ni < 2; ++ni)
            acc[mi][ni] = (v8f){};

    // ---- per-thread staging geometry ----
    // X: 128 rows x 32 cols fp32; each thread: 4 x float4 (rows xrow, xrow+32, ...)
    const int xrow = tid >> 3;
    const int xcol = (tid & 7) << 2;
    // W: 128 rows x 2 groups-of-16; one group per thread
    const int nloc = tid >> 1;
    const int kg   = tid & 1;

    const float* xsrc = X + (size_t)(m0 + xrow) * K_DIM + xcol;
    const int*   qsrc = QW + ((size_t)(n0 + nloc) * KGROUPS + kg) * (GS / 2);
    const float* ssrc = SC + (size_t)(n0 + nloc) * KGROUPS + kg;

    // bias for this wave's two output columns (hoisted out of epilogue)
    float bv[2];
#pragma unroll
    for (int ni = 0; ni < 2; ++ni)
        bv[ni] = BIAS[n0 + wn + ni * 16 + lr];

    // ---- pipeline prologue: load tile 0 into registers ----
    v4f  xr[4];
    v4i  q0, q1;
    float sc;
#pragma unroll
    for (int it = 0; it < 4; ++it)
        xr[it] = *(const v4f*)(xsrc + (size_t)it * 32 * K_DIM);
    q0 = *(const v4i*)(qsrc);
    q1 = *(const v4i*)(qsrc + 4);
    sc = *ssrc;

    for (int kt = 0; kt < NKSTEPS; ++kt) {
        _Float16* Asb = As + (kt & 1) * ATILE;
        _Float16* Bsb = Bs + (kt & 1) * BTILE;

        // ---- convert staged registers -> LDS[buf] ----
#pragma unroll
        for (int it = 0; it < 4; ++it) {
            v4h hx;
            hx.x = (_Float16)xr[it].x; hx.y = (_Float16)xr[it].y;
            hx.z = (_Float16)xr[it].z; hx.w = (_Float16)xr[it].w;
            *(v4h*)(&Asb[(it * 32 + xrow) * LDA + xcol]) = hx;
        }
        {
            const float a2 = sc * (2.0f / 15.0f);
            WBuf w;
#pragma unroll
            for (int j = 0; j < 4; ++j) {
                const int b0 = q0[j], b1 = q1[j];
                w.e[2*j + 0] = (_Float16)fmaf((float)( b0       & 15), a2, -sc);
                w.e[2*j + 1] = (_Float16)fmaf((float)((b0 >> 4) & 15), a2, -sc);
                w.e[2*j + 8] = (_Float16)fmaf((float)( b1       & 15), a2, -sc);
                w.e[2*j + 9] = (_Float16)fmaf((float)((b1 >> 4) & 15), a2, -sc);
            }
            _Float16* dst = &Bsb[nloc * LDB + kg * 16];
            *(v8h*)(dst)     = w.v[0];
            *(v8h*)(dst + 8) = w.v[1];
        }

        __syncthreads();

        // ---- issue next tile's global loads; wait sinks to next iteration ----
        if (kt + 1 < NKSTEPS) {
            const int knext = (kt + 1) * BK;
#pragma unroll
            for (int it = 0; it < 4; ++it)
                xr[it] = *(const v4f*)(xsrc + (size_t)it * 32 * K_DIM + knext);
            qsrc += 2 * (GS / 2);   // two groups per row per K-step
            ssrc += 2;
            q0 = *(const v4i*)(qsrc);
            q1 = *(const v4i*)(qsrc + 4);
            sc = *ssrc;
        }

        // ---- fragments from LDS[buf] + 8 WMMAs ----
        F16x16 afrag[4];
#pragma unroll
        for (int mi = 0; mi < 4; ++mi) {
            const _Float16* base = &Asb[(wm + mi * 16 + lr) * LDA];
            afrag[mi].h[0] = *(const v8h*)(base + 8 * hh);        // K = 8*hh..8*hh+7
            afrag[mi].h[1] = *(const v8h*)(base + 16 + 8 * hh);   // K = 16+8*hh..
        }
        F16x16 bfrag[2];
#pragma unroll
        for (int ni = 0; ni < 2; ++ni) {
            const _Float16* base = &Bsb[(wn + ni * 16 + lr) * LDB];
            bfrag[ni].h[0] = *(const v8h*)(base + 8 * hh);
            bfrag[ni].h[1] = *(const v8h*)(base + 16 + 8 * hh);
        }

#pragma unroll
        for (int mi = 0; mi < 4; ++mi)
#pragma unroll
            for (int ni = 0; ni < 2; ++ni)
                acc[mi][ni] = __builtin_amdgcn_wmma_f32_16x16x32_f16(
                    /*neg_a=*/false, afrag[mi].v,
                    /*neg_b=*/false, bfrag[ni].v,
                    /*c_mod=*/(short)0, acc[mi][ni],
                    /*reuse_a=*/false, /*reuse_b=*/false);
        // single barrier per step is safe with ping-pong buffers:
        // reads of this buf complete (dscnt) before this wave's next barrier.
    }

    // ---- epilogue: bias add + coalesced fp32 stores ----
    // C/D layout: VGPR r: lanes 0-15 -> M=r, lanes 16-31 -> M=8+r; N = lane%16
#pragma unroll
    for (int mi = 0; mi < 4; ++mi) {
        const int mbase = m0 + wm + mi * 16 + 8 * hh;
#pragma unroll
        for (int ni = 0; ni < 2; ++ni) {
            const int n = n0 + wn + ni * 16 + lr;
#pragma unroll
            for (int r = 0; r < 8; ++r)
                OUT[(size_t)(mbase + r) * N_DIM + n] = acc[mi][ni][r] + bv[ni];
        }
    }
}

extern "C" void kernel_launch(void* const* d_in, const int* in_sizes, int n_in,
                              void* d_out, int out_size, void* d_ws, size_t ws_size,
                              hipStream_t stream) {
    const float* X    = (const float*)d_in[0];   // [4,2048,4096] fp32
    const int*   QW   = (const int*)d_in[1];     // [N_GROUPS, 8] int32 (one byte each)
    const float* SC   = (const float*)d_in[2];   // [N_GROUPS, 1] fp32
    const float* BIAS = (const float*)d_in[3];   // [16384] fp32
    float*       OUT  = (float*)d_out;           // [4,2048,16384] fp32

    dim3 grid(N_DIM / BN, M_TOK / BM);           // 128 x 64 blocks
    w4a16_wmma_gemm<<<grid, 256, 0, stream>>>(X, QW, SC, BIAS, OUT);
}
